// LatentNet_20083267076111
// MI455X (gfx1250) — compile-verified, matrix operand order checked by reading, not run on hardware
//
#include <hip/hip_runtime.h>
#include <hip/hip_bf16.h>
#include <math.h>

typedef __attribute__((ext_vector_type(2)))  float   v2f;
typedef __attribute__((ext_vector_type(8)))  float   v8f;
typedef __attribute__((ext_vector_type(16))) __bf16  v16bf;

#define N_ROWS 8192
#define D_IN   60
#define D_PAD  64
#define D_Z    16
#define JT     32                 // j-tile = K of the bf16 WMMA
#define WAVES  4
#define BLOCK  (WAVES * 32)
#define ROWS_PER_WG (WAVES * 16)

__device__ __forceinline__ unsigned short f2bf(float f) {
    unsigned u = __float_as_uint(f);
    unsigned r = ((u >> 16) & 1u) + 0x7FFFu;   // round-to-nearest-even
    return (unsigned short)((u + r) >> 16);
}
__device__ __forceinline__ float fast_rcp(float x) { return __builtin_amdgcn_rcpf(x); }
__device__ __forceinline__ v8f zero8() {
    v8f z;
#pragma unroll
    for (int i = 0; i < 8; ++i) z[i] = 0.f;
    return z;
}

// ---------------------------------------------------------------------------
// Kernel 1: z = fc2(fc1(x)) [N,16], sq[i] = |z_i|^2.  ~40 MFLOP, negligible.
// ---------------------------------------------------------------------------
__global__ void ln_prep(const float* __restrict__ x,
                        const float* __restrict__ w1, const float* __restrict__ b1,
                        const float* __restrict__ w2, const float* __restrict__ b2,
                        float* __restrict__ z, float* __restrict__ sq) {
    int i = blockIdx.x * blockDim.x + threadIdx.x;
    if (i >= N_ROWS) return;
    float xr[D_IN];
#pragma unroll
    for (int d = 0; d < D_IN; ++d) xr[d] = x[i * D_IN + d];
    float n1[32];
    for (int o = 0; o < 32; ++o) {
        float s = b1[o];
#pragma unroll
        for (int d = 0; d < D_IN; ++d) s = fmaf(xr[d], w1[o * D_IN + d], s);
        n1[o] = s;
    }
    float ss = 0.f;
    for (int o = 0; o < 16; ++o) {
        float s = b2[o];
#pragma unroll
        for (int d = 0; d < 32; ++d) s = fmaf(n1[d], w2[o * 32 + d], s);
        z[i * D_Z + o] = s;
        ss = fmaf(s, s, ss);
    }
    sq[i] = ss;
}

// ---------------------------------------------------------------------------
// Kernel 2: fused  logits -> sigmoid -> (rowsum, A@x) -> /rowsum -> fc3 ->
// fc6 -> softmax.  One 16-row tile per wave, stream j in tiles of 32.
// ---------------------------------------------------------------------------
__global__ __launch_bounds__(BLOCK)
void ln_main(const float* __restrict__ x,  const float* __restrict__ z,
             const float* __restrict__ sq,
             const float* __restrict__ w3, const float* __restrict__ b3,
             const float* __restrict__ w6, const float* __restrict__ b6,
             const float* __restrict__ tptr, const float* __restrict__ thptr,
             float* __restrict__ out) {
    __shared__ __align__(16) float          zj[JT][D_Z];                // 2 KB
    __shared__ __align__(16) unsigned short xTj[D_PAD][JT];             // 4 KB  (x^T, bf16)
    __shared__ __align__(16) unsigned short Pl[WAVES][16][JT];          // 4 KB  (sigmoid tile, bf16)
    __shared__ __align__(16) float          hl[WAVES][16][D_PAD];       // 16 KB

    const int tid  = threadIdx.x;
    const int wave = tid >> 5;
    const int lane = tid & 31;
    const int half = lane >> 4;
    const int l16  = lane & 15;
    const int i0   = blockIdx.x * ROWS_PER_WG + wave * 16;

    const float temp  = tptr[0];
    const float theta = thptr[0];

    // z_i A-fragments (fp32 16x16x4 A-layout: lane = M, per-half K pair); loop-invariant.
    v2f aZ[4];
#pragma unroll
    for (int kk = 0; kk < 4; ++kk)
        aZ[kk] = *reinterpret_cast<const v2f*>(z + (i0 + l16) * D_Z + 4 * kk + 2 * half);

    float sqi[8];
#pragma unroll
    for (int r = 0; r < 8; ++r) sqi[r] = sq[i0 + r + 8 * half];

    v8f acc[4];
#pragma unroll
    for (int c = 0; c < 4; ++c) acc[c] = zero8();
    float srow[8];
#pragma unroll
    for (int r = 0; r < 8; ++r) srow[r] = 0.f;

    for (int j0 = 0; j0 < N_ROWS; j0 += JT) {
        __syncthreads();
        // stage z_j (fp32) and transposed bf16 x-tile (cols 60..63 zero-padded)
#pragma unroll
        for (int t = tid; t < JT * D_Z; t += BLOCK) ((float*)zj)[t] = z[j0 * D_Z + t];
#pragma unroll
        for (int t = tid; t < JT * D_PAD; t += BLOCK) {
            int k = t >> 6, n = t & 63;                   // coalesced read of x rows
            float v = (n < D_IN) ? x[(j0 + k) * D_IN + n] : 0.f;
            xTj[n][k] = f2bf(v);
        }
        __syncthreads();

        const float sqj0 = sq[j0 + l16];
        const float sqj1 = sq[j0 + 16 + l16];

        // ---- logits: L = z_i @ z_j^T  (K=16 -> 4 chained fp32 WMMAs per 16x16 tile)
        v8f L0 = zero8(), L1 = zero8();
#pragma unroll
        for (int kk = 0; kk < 4; ++kk) {
            v2f bt0 = *reinterpret_cast<const v2f*>(&zj[l16][4 * kk + 2 * half]);
            v2f bt1 = *reinterpret_cast<const v2f*>(&zj[16 + l16][4 * kk + 2 * half]);
            L0 = __builtin_amdgcn_wmma_f32_16x16x4_f32(false, aZ[kk], false, bt0, (short)0, L0, false, false);
            L1 = __builtin_amdgcn_wmma_f32_16x16x4_f32(false, aZ[kk], false, bt1, (short)0, L1, false, false);
        }

        // ---- dist -> sigmoid -> rowsum partials, spill P tile to LDS in bf16
#pragma unroll
        for (int r = 0; r < 8; ++r) {
            const int gi = i0 + r + 8 * half;
            float d0 = fmaxf(sqi[r] + sqj0 - 2.f * L0[r], 0.f);
            float d1 = fmaxf(sqi[r] + sqj1 - 2.f * L1[r], 0.f);
            float a0 = fast_rcp(1.f + __expf(temp * d0 - theta));   // sigmoid(theta - temp*d)
            float a1 = fast_rcp(1.f + __expf(temp * d1 - theta));
            if (gi == j0 + l16)      a0 = 1.f;                      // fill_diagonal(A, 1)
            if (gi == j0 + 16 + l16) a1 = 1.f;
            srow[r] += a0 + a1;
            Pl[wave][r + 8 * half][l16]      = f2bf(a0);
            Pl[wave][r + 8 * half][16 + l16] = f2bf(a1);
        }
        // in-wave DS ops are in-order: the transposed reads below see the stores above

        // ---- A-fragment of P (bf16 16x32 A-layout) from this wave's private LDS tile
        union { v16bf v; unsigned u[8]; } Af;
        const unsigned* prow = reinterpret_cast<const unsigned*>(&Pl[wave][l16][0]);
#pragma unroll
        for (int vv = 0; vv < 8; ++vv) {
            int K0 = ((vv & 4) << 2) + 8 * half + 2 * (vv & 3);
            Af.u[vv] = prow[K0 >> 1];
        }

        // ---- H += P @ X  (four 16-wide column chunks of the padded 64 columns)
#pragma unroll
        for (int c = 0; c < 4; ++c) {
            union { v16bf v; unsigned u[8]; } Bf;
            const unsigned* xrow = reinterpret_cast<const unsigned*>(&xTj[16 * c + l16][0]);
#pragma unroll
            for (int vv = 0; vv < 8; ++vv) {
                int K0 = ((vv & 4) << 2) + 8 * half + 2 * (vv & 3);
                Bf.u[vv] = xrow[K0 >> 1];
            }
            acc[c] = __builtin_amdgcn_wmma_f32_16x16x32_bf16(false, Af.v, false, Bf.v,
                                                             (short)0, acc[c], false, false);
        }
    }

    // ---- reduce rowsums across the 16 lanes that share each M (xor masks stay in-half)
#pragma unroll
    for (int r = 0; r < 8; ++r) {
#pragma unroll
        for (int m = 1; m < 16; m <<= 1)
            srow[r] += __shfl_xor(srow[r], m, 32);
    }

    // ---- normalize and park h tile in LDS (C-layout -> row-major)
    float rs[8];
#pragma unroll
    for (int r = 0; r < 8; ++r) rs[r] = fast_rcp(srow[r]);   // srow >= 1 (diagonal)
#pragma unroll
    for (int c = 0; c < 4; ++c)
#pragma unroll
        for (int r = 0; r < 8; ++r)
            hl[wave][r + 8 * half][16 * c + l16] = acc[c][r] * rs[r];

    __syncthreads();

    // ---- epilogue: fc3 -> fc6 -> softmax, one row per lane 0..15 of each wave
    if (half == 0) {
        const int M  = l16;
        const int gi = i0 + M;
        float t8[8];
#pragma unroll
        for (int o = 0; o < 8; ++o) {
            float s = b3[o];
#pragma unroll
            for (int d = 0; d < D_IN; ++d) s = fmaf(hl[wave][M][d], w3[o * D_IN + d], s);
            t8[o] = s;
        }
        float lg[10], mx = -1e30f;
#pragma unroll
        for (int c2 = 0; c2 < 10; ++c2) {
            float s = b6[c2];
#pragma unroll
            for (int o = 0; o < 8; ++o) s = fmaf(t8[o], w6[c2 * 8 + o], s);
            lg[c2] = s; mx = fmaxf(mx, s);
        }
        float den = 0.f;
#pragma unroll
        for (int c2 = 0; c2 < 10; ++c2) { lg[c2] = __expf(lg[c2] - mx); den += lg[c2]; }
        float inv = fast_rcp(den);
#pragma unroll
        for (int c2 = 0; c2 < 10; ++c2) out[gi * 10 + c2] = lg[c2] * inv;
    }
}

// ---------------------------------------------------------------------------
extern "C" void kernel_launch(void* const* d_in, const int* in_sizes, int n_in,
                              void* d_out, int out_size, void* d_ws, size_t ws_size,
                              hipStream_t stream) {
    const float* x  = (const float*)d_in[0];
    const float* w1 = (const float*)d_in[1];
    const float* b1 = (const float*)d_in[2];
    const float* w2 = (const float*)d_in[3];
    const float* b2 = (const float*)d_in[4];
    const float* w3 = (const float*)d_in[5];
    const float* b3 = (const float*)d_in[6];
    const float* w6 = (const float*)d_in[7];
    const float* b6 = (const float*)d_in[8];
    const float* tp = (const float*)d_in[9];
    const float* th = (const float*)d_in[10];

    float* z  = (float*)d_ws;                 // [8192,16] fp32
    float* sq = z + N_ROWS * D_Z;             // [8192]    fp32

    ln_prep<<<N_ROWS / 256, 256, 0, stream>>>(x, w1, b1, w2, b2, z, sq);
    ln_main<<<N_ROWS / ROWS_PER_WG, BLOCK, 0, stream>>>(x, z, sq, w3, b3, w6, b6,
                                                        tp, th, (float*)d_out);
}
